// NeuralODE_46454366273769
// MI455X (gfx1250) — matrix operand order repaired and checked by reading, not compiled, not run on hardware
//
#include <hip/hip_runtime.h>

typedef __attribute__((ext_vector_type(16))) _Float16 v16h;
typedef __attribute__((ext_vector_type(8)))  _Float16 v8h;
typedef __attribute__((ext_vector_type(8)))  float    v8f;
typedef __attribute__((ext_vector_type(4)))  float    v4f;

#define T_PTS   32
#define BATCH   2048
#define DDIM    64
#define WDIM    256
#define NSUB    2
#define ROWS    16      // batch rows per block = one WMMA M tile
#define THREADS 512     // 16 wave32 waves
#define NWAVES  16

// DOPRI5 Butcher tableau, flattened [stage][8]: a_s1..a_s5, c_s, pad, pad
static __device__ const float kDopriTab[48] = {
    0.0f,              0.0f,               0.0f,              0.0f,            0.0f,               0.0f,      0.0f, 0.0f,
    0.2f,              0.0f,               0.0f,              0.0f,            0.0f,               0.2f,      0.0f, 0.0f,
    3.0f/40.0f,        9.0f/40.0f,         0.0f,              0.0f,            0.0f,               0.3f,      0.0f, 0.0f,
    44.0f/45.0f,      -56.0f/15.0f,        32.0f/9.0f,        0.0f,            0.0f,               0.8f,      0.0f, 0.0f,
    19372.0f/6561.0f, -25360.0f/2187.0f,   64448.0f/6561.0f, -212.0f/729.0f,   0.0f,               8.0f/9.0f, 0.0f, 0.0f,
    9017.0f/3168.0f,  -355.0f/33.0f,       46732.0f/5247.0f,  49.0f/176.0f,   -5103.0f/18656.0f,   1.0f,      0.0f, 0.0f,
};

// CDNA5 hardware tanh (V_TANH_F32, TRANS32 unit) — single branch-free
// instruction instead of the branchy OCML expansion.
__device__ __forceinline__ float fast_tanh(float x) {
#if __has_builtin(__builtin_amdgcn_tanhf)
    return __builtin_amdgcn_tanhf(x);
#else
    float r;
    asm volatile("v_tanh_f32 %0, %1" : "=v"(r) : "v"(x));
    return r;
#endif
}

__global__ __launch_bounds__(THREADS)
void node_dopri5_kernel(const float* __restrict__ ts,
                        const float* __restrict__ y0,
                        const float* __restrict__ W0,
                        const float* __restrict__ b0,
                        const float* __restrict__ W1,
                        const float* __restrict__ b1,
                        const float* __restrict__ W2,
                        const float* __restrict__ b2,
                        float* __restrict__ out)
{
    // On-chip state for this 16-row batch tile (~62 KB total, << 320 KB/WGP)
    __shared__ __align__(16) float    yS[ROWS * DDIM];        // 4 KB   current y
    __shared__ __align__(16) float    kS[6][ROWS * DDIM];     // 24 KB  k1..k6
    __shared__ __align__(16) float    kpart[4][ROWS * DDIM];  // 16 KB  layer-3 split-K partials
    __shared__ __align__(16) float    coef[6][8];             // 192 B  Butcher tableau
    __shared__ __align__(16) _Float16 xA[ROWS * DDIM];        // 2 KB   vf input (f16)
    __shared__ __align__(16) _Float16 h1S[ROWS * WDIM];       // 8 KB   layer-1 act
    __shared__ __align__(16) _Float16 h2S[ROWS * WDIM];       // 8 KB   layer-2 act

    const int tid      = threadIdx.x;
    const int wave     = tid >> 5;
    const int lane     = tid & 31;
    const int laneHalf = lane >> 4;   // 0: lanes 0-15, 1: lanes 16-31
    const int l16      = lane & 15;
    const int rowBase  = blockIdx.x * ROWS;

    // -------- preload weight B-fragments into registers (f32 -> f16) --------
    // B-matrix (32x16, K x N) fragment k order used by this code (A uses the same):
    //   lanes 0-15 : column n = lane,    K = 0..15 contiguous (2 per VGPR)
    //   lanes 16-31: column n = lane-16, K = 16..31 contiguous
    // B[k][n] == Wt[n][k] because every layer computes x @ Wt^T with Wt stored
    // row-major as (out, in)  ->  each lane reads 16 contiguous floats of row n.
    const int nw   = wave * 16 + l16;        // layers 1/2: output column of this lane
    const int n3   = (wave & 3) * 16 + l16;  // layer 3: output column (N=64)
    const int kgrp = wave >> 2;              // layer 3: K-quarter (split-K over 16 waves)

    auto load_frag = [&](const float* Wrow, int kbase) -> v16h {
        v16h f;
        #pragma unroll
        for (int i = 0; i < 16; i += 4) {
            v4f v = *(const v4f*)(Wrow + kbase + i);
            f[i + 0] = (_Float16)v.x;  f[i + 1] = (_Float16)v.y;
            f[i + 2] = (_Float16)v.z;  f[i + 3] = (_Float16)v.w;
        }
        return f;
    };

    v16h w0f[2];
    #pragma unroll
    for (int kt = 0; kt < 2; ++kt)
        w0f[kt] = load_frag(W0 + nw * DDIM, kt * 32 + laneHalf * 16);

    v16h w1f[8];
    #pragma unroll
    for (int kt = 0; kt < 8; ++kt)
        w1f[kt] = load_frag(W1 + nw * WDIM, kt * 32 + laneHalf * 16);

    v16h w2f[2];   // split-K: this wave covers K-tiles {2*kgrp, 2*kgrp+1}
    #pragma unroll
    for (int j = 0; j < 2; ++j)
        w2f[j] = load_frag(W2 + n3 * WDIM, (kgrp * 2 + j) * 32 + laneHalf * 16);

    const float b0v = b0[nw];
    const float b1v = b1[nw];
    const float b2d = b2[tid & 63];   // bias for this thread's reduction elements
                                      // (e = tid and tid+512 share (e & 63))

    // A-fragment loader from a row-major f16 LDS buffer.
    // 16-bit A (16x32) layout: lane L -> row m = L&15; contiguous
    // [kb..kb+15] with kb = ktile*32 + 16*(L>>4) (matches B fragments).
    auto load_A = [&](const _Float16* buf, int stride, int ktbase) -> v16h {
        const _Float16* p = buf + l16 * stride + ktbase + laneHalf * 16;
        v8h lo = *(const v8h*)(p);
        v8h hi = *(const v8h*)(p + 8);
        v16h a;
        #pragma unroll
        for (int i = 0; i < 8; ++i) { a[i] = lo[i]; a[8 + i] = hi[i]; }
        return a;
    };

    // -------- one-time init: y from y0, out slice t=0, zero kS, tableau -----
    for (int e = tid; e < ROWS * DDIM; e += THREADS) {
        int m = e >> 6, d = e & 63;
        float v = y0[(rowBase + m) * DDIM + d];
        yS[e] = v;
        out[(size_t)(rowBase + m) * DDIM + d] = v;
    }
    // zero kS: stage combinations multiply unused k_j by 0.0; uninitialized
    // LDS could hold NaN bit patterns (0*NaN = NaN), so clear once.
    for (int e = tid; e < 6 * ROWS * DDIM; e += THREADS)
        ((float*)kS)[e] = 0.0f;
    if (tid < 48)
        ((float*)coef)[tid] = kDopriTab[tid];
    __syncthreads();

    // -------- sequential DOPRI5 rollout --------
    for (int iv = 0; iv < T_PTS - 1; ++iv) {
        const float t0    = ts[iv];
        const float hstep = (ts[iv + 1] - t0) * (1.0f / NSUB);

        for (int sub = 0; sub < NSUB; ++sub) {
            const float tc = t0 + hstep * (float)sub;

            for (int s = 0; s < 6; ++s) {
                // ---- stage combination (branch-free, tableau-driven):
                //      ytmp = y + h * sum_j a_sj * k_j  -> xA (f16)
                const float a0 = coef[s][0], a1 = coef[s][1], a2 = coef[s][2];
                const float a3 = coef[s][3], a4 = coef[s][4];
                const float et = __expf(tc + coef[s][5] * hstep);
                for (int e = tid; e < ROWS * DDIM; e += THREADS) {
                    float v = yS[e] + hstep * (a0 * kS[0][e] + a1 * kS[1][e]
                                             + a2 * kS[2][e] + a3 * kS[3][e]
                                             + a4 * kS[4][e]);
                    xA[e] = (_Float16)v;
                }
                __syncthreads();

                v8f acc;

                // ---- layer 1: (16x64) @ (64x256)^tile, N-tile = wave
                {
                    v16h a0f = load_A(xA, DDIM, 0);    // both K-step fragments
                    v16h a1f = load_A(xA, DDIM, 32);   // issued before first WMMA
                    #pragma unroll
                    for (int r = 0; r < 8; ++r) acc[r] = 0.0f;
                    acc = __builtin_amdgcn_wmma_f32_16x16x32_f16(
                        false, a0f, false, w0f[0], (short)0, acc, false, false);
                    acc = __builtin_amdgcn_wmma_f32_16x16x32_f16(
                        false, a1f, false, w0f[1], (short)0, acc, false, false);
                }
                #pragma unroll
                for (int r = 0; r < 8; ++r) {
                    float v = fast_tanh(acc[r] + b0v);
                    h1S[(r + laneHalf * 8) * WDIM + nw] = (_Float16)v;
                }
                __syncthreads();

                // ---- layer 2: (16x256) @ (256x256)^tile  (pipelined A loads)
                {
                    #pragma unroll
                    for (int r = 0; r < 8; ++r) acc[r] = 0.0f;
                    v16h a_cur = load_A(h1S, WDIM, 0);
                    #pragma unroll
                    for (int kt = 0; kt < 8; ++kt) {
                        v16h a_next;
                        if (kt < 7) a_next = load_A(h1S, WDIM, (kt + 1) * 32);
                        acc = __builtin_amdgcn_wmma_f32_16x16x32_f16(
                            false, a_cur, false, w1f[kt], (short)0, acc, false, false);
                        a_cur = a_next;
                    }
                }
                #pragma unroll
                for (int r = 0; r < 8; ++r) {
                    float v = fast_tanh(acc[r] + b1v);
                    h2S[(r + laneHalf * 8) * WDIM + nw] = (_Float16)v;
                }
                __syncthreads();

                // ---- layer 3: (16x256) @ (256x64)^tile, split-K over ALL 16 waves
                //      wave w: N-tile (w&3), K-tiles {2*(w>>2), 2*(w>>2)+1}
                {
                    v16h a0f = load_A(h2S, WDIM, (kgrp * 2 + 0) * 32);
                    v16h a1f = load_A(h2S, WDIM, (kgrp * 2 + 1) * 32);
                    #pragma unroll
                    for (int r = 0; r < 8; ++r) acc[r] = 0.0f;
                    acc = __builtin_amdgcn_wmma_f32_16x16x32_f16(
                        false, a0f, false, w2f[0], (short)0, acc, false, false);
                    acc = __builtin_amdgcn_wmma_f32_16x16x32_f16(
                        false, a1f, false, w2f[1], (short)0, acc, false, false);
                    #pragma unroll
                    for (int r = 0; r < 8; ++r)
                        kpart[kgrp][(r + laneHalf * 8) * DDIM + n3] = acc[r];
                }
                __syncthreads();

                // ---- 4-way split-K reduction + bias + exp(t) scaling -> k_s
                // kS element e is produced & consumed by the same thread, and
                // kpart is not rewritten until after 3 barriers of the next
                // stage -> no trailing barrier needed.
                for (int e = tid; e < ROWS * DDIM; e += THREADS) {
                    float sum = kpart[0][e] + kpart[1][e] + kpart[2][e] + kpart[3][e];
                    kS[s][e] = (sum + b2d) * et;
                }
            } // stages

            // ---- y += h * sum(b_j * k_j)   (kS element-private per thread)
            for (int e = tid; e < ROWS * DDIM; e += THREADS) {
                yS[e] = yS[e] + hstep * ((35.0f/384.0f)   * kS[0][e]
                                       + (500.0f/1113.0f) * kS[2][e]
                                       + (125.0f/192.0f)  * kS[3][e]
                                       + (-2187.0f/6784.0f)* kS[4][e]
                                       + (11.0f/84.0f)    * kS[5][e]);
            }
            __syncthreads();
        } // substeps

        // emit output slice t = iv+1
        for (int e = tid; e < ROWS * DDIM; e += THREADS) {
            int m = e >> 6, d = e & 63;
            out[(size_t)(iv + 1) * (BATCH * DDIM) + (size_t)(rowBase + m) * DDIM + d] = yS[e];
        }
        // yS access pattern is thread-private -> no extra barrier required
    }
}

extern "C" void kernel_launch(void* const* d_in, const int* in_sizes, int n_in,
                              void* d_out, int out_size, void* d_ws, size_t ws_size,
                              hipStream_t stream)
{
    const float* ts = (const float*)d_in[0];
    const float* y0 = (const float*)d_in[1];
    const float* W0 = (const float*)d_in[2];
    const float* b0 = (const float*)d_in[3];
    const float* W1 = (const float*)d_in[4];
    const float* b1 = (const float*)d_in[5];
    const float* W2 = (const float*)d_in[6];
    const float* b2 = (const float*)d_in[7];
    float* out = (float*)d_out;

    dim3 grid(BATCH / ROWS);   // 128 blocks, one 16-row batch tile each
    dim3 block(THREADS);       // 16 wave32 waves
    node_dopri5_kernel<<<grid, block, 0, stream>>>(ts, y0, W0, b0, W1, b1, W2, b2, out);
}
